// SARABrainCore_84086869721661
// MI455X (gfx1250) — matrix operand order repaired and checked by reading, not compiled, not run on hardware
//
#include <hip/hip_runtime.h>

#define TBM 128
#define TBN 128
#define TBK 64
#define BUF_ELEMS (TBM * TBK)        // 8192 u16 per buffer
#define BUF_BYTES (BUF_ELEMS * 2)    // 16384 bytes

typedef unsigned short u16;
typedef __attribute__((ext_vector_type(16))) __bf16 v16bf;
typedef __attribute__((ext_vector_type(8)))  float  v8f;

union FragU { v16bf v; uint4 q[2]; };

__device__ __forceinline__ u16 f32_to_bf16(float f) {
  unsigned u = __float_as_uint(f);
  u += 0x7FFFu + ((u >> 16) & 1u);   // round-to-nearest-even
  return (u16)(u >> 16);
}

__device__ __forceinline__ void zero_acc(v8f acc[4][2]) {
#pragma unroll
  for (int i = 0; i < 4; ++i)
#pragma unroll
    for (int j = 0; j < 2; ++j)
#pragma unroll
      for (int v = 0; v < 8; ++v) acc[i][j][v] = 0.0f;
}

// Async copy of 64 bytes (4 x b128) global -> LDS via the CDNA5 async engine.
// lds is the wave-relative LDS byte address (low 32 bits of the flat pointer).
__device__ __forceinline__ void issue_cp64(unsigned lds, const u16* g) {
#pragma unroll
  for (int i = 0; i < 4; ++i)
    asm volatile("global_load_async_to_lds_b128 %0, %1, off"
                 :: "v"(lds + (unsigned)(i * 16)), "v"(g + i * 8)
                 : "memory");
}

__device__ __forceinline__ void wait_async0() {
  asm volatile("s_wait_asynccnt 0" ::: "memory");
}

// ---------------------------------------------------------------------------
// Core bf16 WMMA GEMM: C(128x128 f32) += A(128xK bf16, row-major, lda)
//                                      x Bt(128xK bf16 rows = B columns, ldb)
// 256 threads = 8 waves; each wave owns a 64x32 tile (4x2 of 16x16).
// Double-buffered LDS, tiles staged with GLOBAL_LOAD_ASYNC_TO_LDS_B128 so the
// copy of tile k+1 overlaps the 16 WMMAs of tile k.
// ---------------------------------------------------------------------------
__device__ __forceinline__ void gemm_block(
    const u16* __restrict__ A, long lda,
    const u16* __restrict__ Bt, long ldb,
    int K, int m0, int n0,
    u16* As, u16* Bs, v8f acc[4][2])
{
  const int tid   = threadIdx.x;
  const int lane  = tid & 31;
  const int w     = tid >> 5;
  const int wm    = w & 1;           // 0..1 (64-row slabs)
  const int wn    = w >> 1;          // 0..3 (32-col slabs)
  const int lrow  = lane & 15;
  const int lhalf = lane >> 4;       // 0/1
  const int ldRow = tid >> 1;        // 0..127
  const int ldOff = (tid & 1) << 5;  // 0/32 elements (64B halves of a row)

  const u16* gA = A  + (size_t)(m0 + ldRow) * lda + ldOff;
  const u16* gB = Bt + (size_t)(n0 + ldRow) * ldb + ldOff;
  const unsigned ldsA0 = (unsigned)(size_t)As + (unsigned)((ldRow * TBK + ldOff) * 2);
  const unsigned ldsB0 = (unsigned)(size_t)Bs + (unsigned)((ldRow * TBK + ldOff) * 2);

  __syncthreads();                   // previous users of LDS are done
  issue_cp64(ldsA0, gA);             // prologue: stage tile 0 into buffer 0
  issue_cp64(ldsB0, gB);

  const int nk = K >> 6;             // K / 64
  for (int it = 0; it < nk; ++it) {
    const int p = it & 1;
    wait_async0();                   // own async writes for tile `it` done
    __syncthreads();                 // everyone's writes visible; prev reads done
    if (it + 1 < nk) {               // stage tile it+1 into the other buffer
      const unsigned q = (unsigned)(1 - p) * BUF_BYTES;
      issue_cp64(ldsA0 + q, gA + (size_t)(it + 1) * TBK);
      issue_cp64(ldsB0 + q, gB + (size_t)(it + 1) * TBK);
    }
    const u16* Ab = As + p * BUF_ELEMS;
    const u16* Bb = Bs + p * BUF_ELEMS;

#pragma unroll
    for (int kk = 0; kk < 2; ++kk) {
      FragU afr[4], bfr[2];
#pragma unroll
      for (int mt = 0; mt < 4; ++mt) {
        // A 16x32 bf16 frag: lanes 0-15 -> K[0..7]+K[16..23], lanes 16-31 -> K[8..15]+K[24..31]
        int rr = (wm * 64 + mt * 16 + lrow) * TBK + kk * 32 + lhalf * 8;
        afr[mt].q[0] = *(const uint4*)&Ab[rr];
        afr[mt].q[1] = *(const uint4*)&Ab[rr + 16];
      }
#pragma unroll
      for (int nt = 0; nt < 2; ++nt) {
        // B 32x16 bf16 frag: lanes 0-15 -> K[0..7]+K[16..23], lanes 16-31 -> K[8..15]+K[24..31], N = lane%16
        int cc = (wn * 32 + nt * 16 + lrow) * TBK + kk * 32 + lhalf * 8;
        bfr[nt].q[0] = *(const uint4*)&Bb[cc];
        bfr[nt].q[1] = *(const uint4*)&Bb[cc + 16];
      }
#pragma unroll
      for (int mt = 0; mt < 4; ++mt)
#pragma unroll
        for (int nt = 0; nt < 2; ++nt)
          acc[mt][nt] = __builtin_amdgcn_wmma_f32_16x16x32_bf16(
              false, afr[mt].v, false, bfr[nt].v, (short)0, acc[mt][nt], false, false);
    }
  }
}

// C/D 16x16 f32 layout: VGPR v, lanes 0-15 -> M=v, lanes 16-31 -> M=v+8; N = lane%16.
#define GEMM_EPILOGUE(ACC, M0, N0, ...)                                        \
  {                                                                            \
    const int _lane = threadIdx.x & 31;                                        \
    const int _w = threadIdx.x >> 5;                                           \
    const int _wm = _w & 1, _wn = _w >> 1;                                     \
    const int _lrow = _lane & 15, _lhalf = _lane >> 4;                         \
    _Pragma("unroll") for (int mt = 0; mt < 4; ++mt)                           \
    _Pragma("unroll") for (int nt = 0; nt < 2; ++nt)                           \
    _Pragma("unroll") for (int v = 0; v < 8; ++v) {                            \
      const int r = (M0) + _wm * 64 + mt * 16 + _lhalf * 8 + v;                \
      const int c = (N0) + _wn * 32 + nt * 16 + _lrow;                         \
      const float accv = ACC[mt][nt][v];                                       \
      __VA_ARGS__                                                              \
    }                                                                          \
  }

// ------------------------------- prep kernels ------------------------------

__global__ void k_f32_to_bf16(const float* __restrict__ s, u16* __restrict__ d, int n) {
  int i = blockIdx.x * 256 + threadIdx.x;
  if (i < n) d[i] = f32_to_bf16(s[i]);
}

__global__ void k_zero(float* __restrict__ p, int n) {
  int i = blockIdx.x * 256 + threadIdx.x;
  if (i < n) p[i] = 0.0f;
}

__global__ void k_copy(const float* __restrict__ s, float* __restrict__ d, int n) {
  int i = blockIdx.x * 256 + threadIdx.x;
  if (i < n) d[i] = s[i];
}

// Wt[n*K + k] = bf16(W[k*N + n]) : 32x32 LDS-tiled transpose + downconvert
__global__ void k_transpose_bf16(const float* __restrict__ W, u16* __restrict__ Wt,
                                 int K, int N) {
  __shared__ float tile[32][33];
  int k0 = blockIdx.x * 32, n0 = blockIdx.y * 32;
  int tx = threadIdx.x & 31, ty = threadIdx.x >> 5; // ty: 0..7
#pragma unroll
  for (int i = 0; i < 4; ++i) {
    int ky = ty + i * 8;
    tile[ky][tx] = W[(size_t)(k0 + ky) * N + n0 + tx];
  }
  __syncthreads();
#pragma unroll
  for (int i = 0; i < 4; ++i) {
    int ny = ty + i * 8;
    Wt[(size_t)(n0 + ny) * K + k0 + tx] = f32_to_bf16(tile[tx][ny]);
  }
}

// LayerNorm over H=2048; also emits bf16(ctx + rec) for the gate GEMM.
__global__ void k_ln(const float* __restrict__ rec, const float* __restrict__ gamma,
                     const float* __restrict__ beta, float* __restrict__ ctx,
                     u16* __restrict__ tsum) {
  const int row = blockIdx.x, tid = threadIdx.x;
  const float* rp = rec + (size_t)row * 2048;
  float vals[8], s = 0.f, s2 = 0.f;
#pragma unroll
  for (int i = 0; i < 8; ++i) {
    float v = rp[tid + 256 * i];
    vals[i] = v; s += v; s2 += v * v;
  }
  __shared__ float sh[512];
  sh[tid] = s; sh[256 + tid] = s2;
  __syncthreads();
  for (int o = 128; o > 0; o >>= 1) {
    if (tid < o) { sh[tid] += sh[tid + o]; sh[256 + tid] += sh[256 + tid + o]; }
    __syncthreads();
  }
  float mean = sh[0] * (1.0f / 2048.0f);
  float var  = sh[256] * (1.0f / 2048.0f) - mean * mean;
  float inv  = rsqrtf(var + 1e-5f);
#pragma unroll
  for (int i = 0; i < 8; ++i) {
    int c = tid + 256 * i;
    float v = vals[i];
    float cx = (v - mean) * inv * gamma[c] + beta[c];
    size_t idx = (size_t)row * 2048 + c;
    ctx[idx]  = cx;
    tsum[idx] = f32_to_bf16(cx + v);
  }
}

__global__ void k_feat(const float* __restrict__ spk, u16* __restrict__ feat, int n) {
  int i = blockIdx.x * 256 + threadIdx.x;
  if (i < n) feat[i] = f32_to_bf16(spk[i] * 0.0625f); // mean over T=16
}

// ------------------------------- GEMM kernels ------------------------------

__global__ void __launch_bounds__(256)
k_gemm_proj(const u16* __restrict__ x16, const u16* __restrict__ Wpt,
            const float* __restrict__ bp, u16* __restrict__ inp16) {
  __shared__ u16 As[2 * BUF_ELEMS], Bs[2 * BUF_ELEMS];
  v8f acc[4][2]; zero_acc(acc);
  int m0 = blockIdx.y * TBM, n0 = blockIdx.x * TBN;
  gemm_block(x16, 1024, Wpt, 1024, 1024, m0, n0, As, Bs, acc);
  GEMM_EPILOGUE(acc, m0, n0,
    inp16[(size_t)r * 2048 + c] = f32_to_bf16(accv + bp[c]);
  )
}

__global__ void __launch_bounds__(256)
k_gemm_fc(const u16* __restrict__ inpT, const u16* __restrict__ Wfct,
          const float* __restrict__ bfc, float* __restrict__ rec) {
  __shared__ u16 As[2 * BUF_ELEMS], Bs[2 * BUF_ELEMS];
  v8f acc[4][2]; zero_acc(acc);
  int m0 = blockIdx.y * TBM, n0 = blockIdx.x * TBN;
  gemm_block(inpT, (long)16 * 2048, Wfct, 2048, 2048, m0, n0, As, Bs, acc);
  GEMM_EPILOGUE(acc, m0, n0,
    size_t idx = (size_t)r * 2048 + c;
    float u = tanhf(accv + bfc[c]);
    rec[idx] = 0.9f * rec[idx] + 0.1f * u;
  )
}

__global__ void __launch_bounds__(256)
k_gemm_gate(const u16* __restrict__ tsum, const u16* __restrict__ Wgt,
            const float* __restrict__ bg, const float* __restrict__ ctx,
            const float* __restrict__ rec, u16* __restrict__ gated) {
  __shared__ u16 As[2 * BUF_ELEMS], Bs[2 * BUF_ELEMS];
  v8f acc[4][2]; zero_acc(acc);
  int m0 = blockIdx.y * TBM, n0 = blockIdx.x * TBN;
  gemm_block(tsum, 2048, Wgt, 2048, 2048, m0, n0, As, Bs, acc);
  GEMM_EPILOGUE(acc, m0, n0,
    size_t idx = (size_t)r * 2048 + c;
    float g = 1.0f / (1.0f + expf(-(accv + bg[c])));
    gated[idx] = f32_to_bf16(g * ctx[idx] + (1.0f - g) * rec[idx]);
  )
}

__global__ void __launch_bounds__(256)
k_gemm_attr(const u16* __restrict__ gated, const u16* __restrict__ Wenct,
            const u16* __restrict__ attrIn, const u16* __restrict__ Wrect,
            const float* __restrict__ benc, const float* __restrict__ brec,
            float* __restrict__ attr, u16* __restrict__ attrOut,
            float* __restrict__ spk) {
  __shared__ u16 As[2 * BUF_ELEMS], Bs[2 * BUF_ELEMS];
  v8f acc[4][2]; zero_acc(acc);
  int m0 = blockIdx.y * TBM, n0 = blockIdx.x * TBN;
  gemm_block(gated,  2048, Wenct, 2048, 2048, m0, n0, As, Bs, acc);
  gemm_block(attrIn, 2048, Wrect, 2048, 2048, m0, n0, As, Bs, acc);
  GEMM_EPILOGUE(acc, m0, n0,
    size_t idx = (size_t)r * 2048 + c;
    float cur = accv + benc[c] + brec[c];
    float a = attr[idx];
    float nv = a + (cur - a) * 0.5f;       // TAU = 2
    float s = nv > 1.0f ? 1.0f : 0.0f;     // THR = 1
    spk[idx] += s;
    float na = nv - s;
    attr[idx] = na;
    attrOut[idx] = f32_to_bf16(na);
  )
}

__global__ void __launch_bounds__(256)
k_gemm_ro(const u16* __restrict__ feat, const u16* __restrict__ Wrot,
          const float* __restrict__ bro, float* __restrict__ out) {
  __shared__ u16 As[2 * BUF_ELEMS], Bs[2 * BUF_ELEMS];
  v8f acc[4][2]; zero_acc(acc);
  int m0 = blockIdx.y * TBM, n0 = blockIdx.x * TBN;
  gemm_block(feat, 2048, Wrot, 2048, 2048, m0, n0, As, Bs, acc);
  GEMM_EPILOGUE(acc, m0, n0,
    out[(size_t)r * 1024 + c] = accv + bro[c];
  )
}

// ------------------------------- host driver -------------------------------

extern "C" void kernel_launch(void* const* d_in, const int* in_sizes, int n_in,
                              void* d_out, int out_size, void* d_ws, size_t ws_size,
                              hipStream_t stream) {
  (void)in_sizes; (void)n_in; (void)out_size; (void)ws_size;
  const float* x     = (const float*)d_in[0];
  const float* Wp    = (const float*)d_in[1];
  const float* bp    = (const float*)d_in[2];
  const float* Wfc   = (const float*)d_in[3];
  const float* bfc   = (const float*)d_in[4];
  const float* gamma = (const float*)d_in[5];
  const float* betaL = (const float*)d_in[6];
  const float* Wg    = (const float*)d_in[7];
  const float* bg    = (const float*)d_in[8];
  const float* Wenc  = (const float*)d_in[9];
  const float* benc  = (const float*)d_in[10];
  const float* Wrec  = (const float*)d_in[11];
  const float* brec  = (const float*)d_in[12];
  const float* Wro   = (const float*)d_in[13];
  const float* bro   = (const float*)d_in[14];

  const int B = 1024, T = 16, D = 1024, H = 2048, O = 1024;

  char* p = (char*)d_ws;
  auto take = [&](size_t bytes) -> char* {
    char* r = p;
    p += (bytes + 255) & ~(size_t)255;
    return r;
  };
  u16*  x16   = (u16*)  take((size_t)B * T * D * 2);
  u16*  inp16 = (u16*)  take((size_t)B * T * H * 2);
  u16*  Wpt   = (u16*)  take((size_t)D * H * 2);
  u16*  Wfct  = (u16*)  take((size_t)H * H * 2);
  u16*  Wgt   = (u16*)  take((size_t)H * H * 2);
  u16*  Wenct = (u16*)  take((size_t)H * H * 2);
  u16*  Wrect = (u16*)  take((size_t)H * H * 2);
  u16*  Wrot  = (u16*)  take((size_t)H * O * 2);
  float* rec  = (float*)take((size_t)B * H * 4);
  float* ctx  = (float*)take((size_t)B * H * 4);
  float* attr = (float*)take((size_t)B * H * 4);
  float* spk  = (float*)take((size_t)B * H * 4);
  u16*  tsum  = (u16*)  take((size_t)B * H * 2);
  u16*  gated = (u16*)  take((size_t)B * H * 2);
  u16*  attrA = (u16*)  take((size_t)B * H * 2);
  u16*  attrB = (u16*)  take((size_t)B * H * 2);
  u16*  feat  = (u16*)  take((size_t)B * H * 2);

  // One-time conversions (deterministic, re-done every launch)
  {
    int n = B * T * D;
    k_f32_to_bf16<<<n / 256, 256, 0, stream>>>(x, x16, n);
  }
  k_transpose_bf16<<<dim3(D / 32, H / 32), 256, 0, stream>>>(Wp,   Wpt,   D, H);
  k_transpose_bf16<<<dim3(H / 32, H / 32), 256, 0, stream>>>(Wfc,  Wfct,  H, H);
  k_transpose_bf16<<<dim3(H / 32, H / 32), 256, 0, stream>>>(Wg,   Wgt,   H, H);
  k_transpose_bf16<<<dim3(H / 32, H / 32), 256, 0, stream>>>(Wenc, Wenct, H, H);
  k_transpose_bf16<<<dim3(H / 32, H / 32), 256, 0, stream>>>(Wrec, Wrect, H, H);
  k_transpose_bf16<<<dim3(H / 32, O / 32), 256, 0, stream>>>(Wro,  Wrot,  H, O);

  k_zero<<<(B * H) / 256, 256, 0, stream>>>(rec,  B * H);
  k_zero<<<(B * H) / 256, 256, 0, stream>>>(attr, B * H);
  k_zero<<<(B * H) / 256, 256, 0, stream>>>(spk,  B * H);
  k_zero<<<(B * H / 2) / 256, 256, 0, stream>>>((float*)attrA, B * H / 2);

  // Hoisted input projection: [B*T, D] @ [D, H] -> bf16 [B, T, H]
  k_gemm_proj<<<dim3(H / TBN, (B * T) / TBM), 256, 0, stream>>>(x16, Wpt, bp, inp16);

  const dim3 gstep(H / TBN, B / TBM);
  for (int t = 0; t < T; ++t) {
    k_gemm_fc<<<gstep, 256, 0, stream>>>(inp16 + (size_t)t * H, Wfct, bfc, rec);
    k_ln<<<B, 256, 0, stream>>>(rec, gamma, betaL, ctx, tsum);
    k_gemm_gate<<<gstep, 256, 0, stream>>>(tsum, Wgt, bg, ctx, rec, gated);
    const u16* ain = (t & 1) ? attrB : attrA;
    u16* aout      = (t & 1) ? attrA : attrB;
    k_gemm_attr<<<gstep, 256, 0, stream>>>(gated, Wenct, ain, Wrect,
                                           benc, brec, attr, aout, spk);
  }

  k_feat<<<(B * H) / 256, 256, 0, stream>>>(spk, feat, B * H);
  k_gemm_ro<<<dim3(O / TBN, B / TBM), 256, 0, stream>>>(feat, Wrot, bro, (float*)d_out);

  // Tuple outputs: (out, rec, attr)
  k_copy<<<(B * H) / 256, 256, 0, stream>>>(rec,  (float*)d_out + (size_t)B * O, B * H);
  k_copy<<<(B * H) / 256, 256, 0, stream>>>(attr, (float*)d_out + (size_t)B * O + (size_t)B * H, B * H);
}
